// Global_Positional_Relational_Block_47991964565565
// MI455X (gfx1250) — matrix operand order repaired and checked by reading, not compiled, not run on hardware
//
#include <hip/hip_runtime.h>
#include <hip/hip_bf16.h>

// Shapes (fixed by the reference): B=2, N=2048, C=1024, H=16, D=64
#define HH     16
#define NSEQ   2048
#define CDIM   1024
#define DHEAD  64
#define BB     2

typedef __attribute__((ext_vector_type(16))) __bf16 v16bf;
typedef __attribute__((ext_vector_type(8)))  float  v8f;

__device__ __forceinline__ unsigned short f2bf(float f) {
  unsigned int u = __float_as_uint(f);
  u += 0x7fffu + ((u >> 16) & 1u);           // round-to-nearest-even
  return (unsigned short)(u >> 16);
}
__device__ __forceinline__ float bf2f(unsigned short s) {
  return __uint_as_float(((unsigned int)s) << 16);
}

union Frag16 { uint4 u[2]; v16bf v; };

// A-fragment (16x32 bf16): row = lane&15; 16 contiguous K elems at kb=(lane>>4)*16.
// Caller passes pointer to (row base + k + kb).
__device__ __forceinline__ v16bf load_afrag(const unsigned short* p) {
  Frag16 f;
  f.u[0] = *reinterpret_cast<const uint4*>(p);
  f.u[1] = *reinterpret_cast<const uint4*>(p + 8);
  return f.v;
}
// B-fragment (32x16 bf16): col = lane&15; 16 contiguous K elems at kb=(lane>>4)*16.
// Caller passes pointer to (col base + k + kb).
__device__ __forceinline__ v16bf load_bfrag(const unsigned short* p) {
  Frag16 f;
  f.u[0] = *reinterpret_cast<const uint4*>(p);
  f.u[1] = *reinterpret_cast<const uint4*>(p + 8);
  return f.v;
}
__device__ __forceinline__ v8f wmma_bf16(v16bf a, v16bf b, v8f c) {
  return __builtin_amdgcn_wmma_f32_16x16x32_bf16(false, a, false, b, (short)0, c, false, false);
}

// ---------------- fp32 -> bf16 elementwise convert ----------------
__global__ void conv_k(const float* __restrict__ src, unsigned short* __restrict__ dst, int n) {
  int i = blockIdx.x * blockDim.x + threadIdx.x;
  if (i < n) dst[i] = f2bf(src[i]);
}

// ------- fp32 [K,N] -> bf16 [N,K] transpose-convert (LDS tiled) -------
__global__ __launch_bounds__(256) void convT_k(const float* __restrict__ src,
                                               unsigned short* __restrict__ dst,
                                               int K, int N) {
  __shared__ float tile[32][33];
  int kt = blockIdx.y * 32, nt = blockIdx.x * 32;
  for (int r = threadIdx.y; r < 32; r += 8)
    tile[r][threadIdx.x] = src[(size_t)(kt + r) * N + nt + threadIdx.x];
  __syncthreads();
  for (int r = threadIdx.y; r < 32; r += 8)
    dst[(size_t)(nt + r) * K + kt + threadIdx.x] = f2bf(tile[threadIdx.x][r]);
}

// ---------------- generic bf16 WMMA GEMM, K=1024, fused epilogues ----------------
// A: [4096, 1024] bf16 row-major.  Bt: [Ncols, 1024] bf16 (pre-transposed weights).
// mode 0: write q [B,H,N,D] bf16.  mode 1: write k,v [B,H,N,D] + vT [B,H,D,N] bf16.
// mode 2: write fp32 [4096, 1024] (+bias) to outF.
__global__ __launch_bounds__(256) void gemm_bf16_k(
    const unsigned short* __restrict__ A,
    const unsigned short* __restrict__ Bt,
    const float* __restrict__ bias,
    int mode,
    unsigned short* __restrict__ outQ,
    unsigned short* __restrict__ outK,
    unsigned short* __restrict__ outV,
    unsigned short* __restrict__ outVT,
    float* __restrict__ outF) {
  const int lane = threadIdx.x & 31;
  const int w  = threadIdx.x >> 5;
  const int wm = w & 3, wn = w >> 2;          // 4 M-tiles x 2 N-strips per block
  const int i0 = blockIdx.x * 64 + wm * 16;   // M block = 64 rows
  const int n0 = blockIdx.y * 128 + wn * 64;  // N block = 128 cols
  const int hl = lane & 15;
  const int kbA = (lane >> 4) * 16;
  const int kbB = (lane >> 4) * 16;
  const int row = i0 + hl;

  v8f acc[4] = {};
  for (int k = 0; k < CDIM; k += 32) {
    v16bf a = load_afrag(A + (size_t)row * CDIM + k + kbA);
#pragma unroll
    for (int t = 0; t < 4; ++t) {
      int col = n0 + t * 16 + hl;
      v16bf b = load_bfrag(Bt + (size_t)col * CDIM + k + kbB);
      acc[t] = wmma_bf16(a, b, acc[t]);
    }
  }
#pragma unroll
  for (int t = 0; t < 4; ++t) {
    int col = n0 + t * 16 + hl;
    float bv = bias[col];
#pragma unroll
    for (int g = 0; g < 8; ++g) {
      int r = i0 + g + ((lane >> 4) << 3);    // C layout: row = g + (lane>=16 ? 8 : 0)
      int rb = r >> 11;                        // batch
      int n  = r & (NSEQ - 1);
      float val = acc[t][g] + bv;
      if (mode == 0) {
        int h = col >> 6, d = col & 63;
        outQ[(((size_t)rb * HH + h) * NSEQ + n) * DHEAD + d] = f2bf(val);
      } else if (mode == 1) {
        int s = col >> 10, rem = col & 1023;
        int h = rem >> 6, d = rem & 63;
        if (s == 0) {
          outK[(((size_t)rb * HH + h) * NSEQ + n) * DHEAD + d] = f2bf(val);
        } else {
          unsigned short bvv = f2bf(val);
          outV[(((size_t)rb * HH + h) * NSEQ + n) * DHEAD + d] = bvv;
          outVT[(((size_t)rb * HH + h) * DHEAD + d) * NSEQ + n] = bvv;
        }
      } else {
        outF[(size_t)r * CDIM + col] = val;
      }
    }
  }
}

// ---------------- zero the j=i+1 diagonal of Srel ----------------
__global__ void diag_zero_k(unsigned short* __restrict__ srel) {
  size_t base = (size_t)blockIdx.x * NSEQ * NSEQ;
  for (int r = threadIdx.x; r < NSEQ - 1; r += blockDim.x)
    srel[base + (size_t)r * NSEQ + r + 1] = 0;
}

// ---------------- QEr = q @ Er^T with fused skew-scatter into Srel ----------------
// QEr[i,m]  ->  Srel[i, m+i-N+1] if m >= N-1-i,  else Srel[i-1, m+i+1] (i>=1).
__global__ __launch_bounds__(256) void qer_skew_k(
    const unsigned short* __restrict__ q,    // [B*H, N, D]
    const unsigned short* __restrict__ Erb,  // [N, D]
    unsigned short* __restrict__ srel) {     // [B*H, N, N]
  const int lane = threadIdx.x & 31;
  const int w  = threadIdx.x >> 5;
  const int bh = blockIdx.z;
  const int i0 = blockIdx.y * 16;
  const int m0 = blockIdx.x * 512 + w * 64;
  const int hl = lane & 15;
  const int kbA = (lane >> 4) * 16;
  const int kbB = (lane >> 4) * 16;
  const unsigned short* qh = q + (size_t)bh * NSEQ * DHEAD;

  v16bf a0 = load_afrag(qh + (size_t)(i0 + hl) * DHEAD + 0  + kbA);
  v16bf a1 = load_afrag(qh + (size_t)(i0 + hl) * DHEAD + 32 + kbA);
  v8f acc[4] = {};
#pragma unroll
  for (int t = 0; t < 4; ++t) {
    int m = m0 + t * 16 + hl;
    v16bf b0 = load_bfrag(Erb + (size_t)m * DHEAD + 0  + kbB);
    v16bf b1 = load_bfrag(Erb + (size_t)m * DHEAD + 32 + kbB);
    acc[t] = wmma_bf16(a0, b0, acc[t]);
    acc[t] = wmma_bf16(a1, b1, acc[t]);
  }
  unsigned short* sh = srel + (size_t)bh * NSEQ * NSEQ;
#pragma unroll
  for (int t = 0; t < 4; ++t) {
    int m = m0 + t * 16 + hl;
#pragma unroll
    for (int g = 0; g < 8; ++g) {
      int i = i0 + g + ((lane >> 4) << 3);
      unsigned short v = f2bf(acc[t][g]);
      if (m >= NSEQ - 1 - i) {
        sh[(size_t)i * NSEQ + (m + i - (NSEQ - 1))] = v;
      } else if (i >= 1) {
        sh[(size_t)(i - 1) * NSEQ + (m + i + 1)] = v;
      }
    }
  }
}

// ---------------- flash attention with Srel bias ----------------
// grid (N/64, B*H), block 128 (4 waves x 16 rows). attn=(q.kT+Srel)*0.125, softmax, @v.
__global__ __launch_bounds__(128) void flash_k(
    const unsigned short* __restrict__ q,     // [B*H, N, D]
    const unsigned short* __restrict__ kk,    // [B*H, N, D]
    const unsigned short* __restrict__ vT,    // [B*H, D, N]
    const unsigned short* __restrict__ srel,  // [B*H, N, N]
    unsigned short* __restrict__ outA) {      // [B, N, C] bf16
  __shared__ __align__(16) unsigned short Pl[4][16 * 64];
  const int lane = threadIdx.x & 31;
  const int w  = threadIdx.x >> 5;
  const int bh = blockIdx.y;
  const int b = bh >> 4, h = bh & 15;
  const int i0 = blockIdx.x * 64 + w * 16;
  const int hl = lane & 15;
  const int kbA = (lane >> 4) * 16;
  const int kbB = (lane >> 4) * 16;
  const unsigned short* qh  = q    + (size_t)bh * NSEQ * DHEAD;
  const unsigned short* kh  = kk   + (size_t)bh * NSEQ * DHEAD;
  const unsigned short* vTh = vT   + (size_t)bh * DHEAD * NSEQ;
  const unsigned short* sh  = srel + (size_t)bh * NSEQ * NSEQ;
  unsigned short* Pw = &Pl[w][0];

  v16bf qa0 = load_afrag(qh + (size_t)(i0 + hl) * DHEAD + 0  + kbA);
  v16bf qa1 = load_afrag(qh + (size_t)(i0 + hl) * DHEAD + 32 + kbA);

  float mrow[8], lrow[8];
  v8f o[4] = {};
#pragma unroll
  for (int g = 0; g < 8; ++g) { mrow[g] = -1e30f; lrow[g] = 0.f; }

  for (int j0 = 0; j0 < NSEQ; j0 += 64) {
    // S tile 16x64 = q @ k^T
    v8f s[4];
#pragma unroll
    for (int t = 0; t < 4; ++t) {
      const unsigned short* kb_ = kh + (size_t)(j0 + t * 16 + hl) * DHEAD;
      v16bf b0 = load_bfrag(kb_ + 0  + kbB);
      v16bf b1 = load_bfrag(kb_ + 32 + kbB);
      v8f a = {};
      a = wmma_bf16(qa0, b0, a);
      a = wmma_bf16(qa1, b1, a);
      s[t] = a;
    }
    // + Srel, * scale
#pragma unroll
    for (int t = 0; t < 4; ++t) {
#pragma unroll
      for (int g = 0; g < 8; ++g) {
        int i = i0 + g + ((lane >> 4) << 3);
        float sr = bf2f(sh[(size_t)i * NSEQ + j0 + t * 16 + hl]);
        s[t][g] = (s[t][g] + sr) * 0.125f;
      }
    }
    // online softmax (rows split across 16-lane halves; reduce with xor masks 1..8)
    float corr[8];
#pragma unroll
    for (int g = 0; g < 8; ++g) {
      float mx = fmaxf(fmaxf(s[0][g], s[1][g]), fmaxf(s[2][g], s[3][g]));
#pragma unroll
      for (int off = 1; off < 16; off <<= 1) mx = fmaxf(mx, __shfl_xor(mx, off, 32));
      float nm = fmaxf(mrow[g], mx);
      corr[g] = __expf(mrow[g] - nm);
      mrow[g] = nm;
    }
    float psum[8];
#pragma unroll
    for (int g = 0; g < 8; ++g) psum[g] = 0.f;
#pragma unroll
    for (int t = 0; t < 4; ++t) {
#pragma unroll
      for (int g = 0; g < 8; ++g) {
        float p = __expf(s[t][g] - mrow[g]);
        s[t][g] = p;
        psum[g] += p;
      }
    }
#pragma unroll
    for (int g = 0; g < 8; ++g) {
      float ps = psum[g];
#pragma unroll
      for (int off = 1; off < 16; off <<= 1) ps += __shfl_xor(ps, off, 32);
      lrow[g] = lrow[g] * corr[g] + ps;
#pragma unroll
      for (int dt = 0; dt < 4; ++dt) o[dt][g] *= corr[g];
    }
    // stage P (C layout -> A layout) through wave-private LDS
    asm volatile("s_wait_dscnt 0" ::: "memory");   // prior iter's ds loads done
#pragma unroll
    for (int t = 0; t < 4; ++t)
#pragma unroll
      for (int g = 0; g < 8; ++g) {
        int rr = g + ((lane >> 4) << 3);
        Pw[rr * 64 + t * 16 + hl] = f2bf(s[t][g]);
      }
    asm volatile("s_wait_dscnt 0" ::: "memory");   // stores visible to whole wave
    v16bf pa0 = load_afrag(Pw + (size_t)hl * 64 + 0  + kbA);
    v16bf pa1 = load_afrag(Pw + (size_t)hl * 64 + 32 + kbA);
    // O += P @ V   (B frags from vT: col=d, contiguous j)
#pragma unroll
    for (int dt = 0; dt < 4; ++dt) {
      const unsigned short* vb = vTh + (size_t)(dt * 16 + hl) * NSEQ + j0;
      v16bf b0 = load_bfrag(vb + 0  + kbB);
      v16bf b1 = load_bfrag(vb + 32 + kbB);
      o[dt] = wmma_bf16(pa0, b0, o[dt]);
      o[dt] = wmma_bf16(pa1, b1, o[dt]);
    }
  }
  // normalize and write [B,N,C] bf16
#pragma unroll
  for (int g = 0; g < 8; ++g) {
    float inv = 1.0f / lrow[g];
    int n = i0 + g + ((lane >> 4) << 3);
#pragma unroll
    for (int dt = 0; dt < 4; ++dt) {
      outA[((size_t)b * NSEQ + n) * CDIM + h * DHEAD + dt * 16 + hl] = f2bf(o[dt][g] * inv);
    }
  }
}

extern "C" void kernel_launch(void* const* d_in, const int* in_sizes, int n_in,
                              void* d_out, int out_size, void* d_ws, size_t ws_size,
                              hipStream_t stream) {
  (void)in_sizes; (void)n_in; (void)out_size; (void)ws_size;
  const float* x    = (const float*)d_in[0];
  const float* q_w  = (const float*)d_in[1];
  const float* q_b  = (const float*)d_in[2];
  const float* kv_w = (const float*)d_in[3];
  const float* kv_b = (const float*)d_in[4];
  const float* pj_w = (const float*)d_in[5];
  const float* pj_b = (const float*)d_in[6];
  const float* Er   = (const float*)d_in[7];

  char* ws = (char*)d_ws;
  size_t off = 0;
  auto take = [&](size_t bytes) -> void* {
    void* p = ws + off;
    off += (bytes + 255) & ~size_t(255);
    return p;
  };
  const size_t M = (size_t)BB * NSEQ;                            // 4096
  unsigned short* xb    = (unsigned short*)take(M * CDIM * 2);            // 8 MiB
  unsigned short* Erb   = (unsigned short*)take((size_t)NSEQ * DHEAD * 2);// 256 KiB
  unsigned short* qwT   = (unsigned short*)take((size_t)CDIM * CDIM * 2); // 2 MiB
  unsigned short* kvwT  = (unsigned short*)take((size_t)2 * CDIM * CDIM * 2); // 4 MiB
  unsigned short* pwT   = (unsigned short*)take((size_t)CDIM * CDIM * 2); // 2 MiB
  unsigned short* qb    = (unsigned short*)take(M * CDIM * 2);            // 8 MiB
  unsigned short* kb    = (unsigned short*)take(M * CDIM * 2);            // 8 MiB
  unsigned short* vb    = (unsigned short*)take(M * CDIM * 2);            // 8 MiB
  unsigned short* vTb   = (unsigned short*)take(M * CDIM * 2);            // 8 MiB
  unsigned short* attnb = (unsigned short*)take(M * CDIM * 2);            // 8 MiB
  unsigned short* srel  = (unsigned short*)take((size_t)BB * HH * NSEQ * NSEQ * 2); // 256 MiB

  // 1) converts
  {
    int n = (int)(M * CDIM);
    conv_k<<<(n + 255) / 256, 256, 0, stream>>>(x, xb, n);
    int ne = NSEQ * DHEAD;
    conv_k<<<(ne + 255) / 256, 256, 0, stream>>>(Er, Erb, ne);
    convT_k<<<dim3(CDIM / 32, CDIM / 32), dim3(32, 8), 0, stream>>>(q_w, qwT, CDIM, CDIM);
    convT_k<<<dim3(2 * CDIM / 32, CDIM / 32), dim3(32, 8), 0, stream>>>(kv_w, kvwT, CDIM, 2 * CDIM);
    convT_k<<<dim3(CDIM / 32, CDIM / 32), dim3(32, 8), 0, stream>>>(pj_w, pwT, CDIM, CDIM);
  }
  // 2) q = x@q_w + q_b  -> [B,H,N,D] bf16
  gemm_bf16_k<<<dim3(M / 64, CDIM / 128), 256, 0, stream>>>(
      xb, qwT, q_b, 0, qb, nullptr, nullptr, nullptr, nullptr);
  // 3) kv = x@kv_w + kv_b -> k,v,[vT] bf16
  gemm_bf16_k<<<dim3(M / 64, 2 * CDIM / 128), 256, 0, stream>>>(
      xb, kvwT, kv_b, 1, nullptr, kb, vb, vTb, nullptr);
  // 4) Srel: zero diagonal, then QEr GEMM with fused skew scatter
  diag_zero_k<<<BB * HH, 256, 0, stream>>>(srel);
  qer_skew_k<<<dim3(NSEQ / 512, NSEQ / 16, BB * HH), 256, 0, stream>>>(qb, Erb, srel);
  // 5) flash attention with Srel bias -> attnb [B,N,C] bf16
  flash_k<<<dim3(NSEQ / 64, BB * HH), 128, 0, stream>>>(qb, kb, vTb, srel, attnb);
  // 6) out = attn @ proj_w + proj_b -> fp32 d_out
  gemm_bf16_k<<<dim3(M / 64, CDIM / 128), 256, 0, stream>>>(
      attnb, pwT, pj_b, 2, nullptr, nullptr, nullptr, nullptr, (float*)d_out);
}
